// S2P2Encoder_41351945126103
// MI455X (gfx1250) — matrix operand order repaired and checked by reading, not compiled, not run on hardware
//
#include <hip/hip_runtime.h>
#include <hip/hip_bf16.h>
#include <math.h>

// ---------------------------------------------------------------------------
// Types for CDNA5 WMMA
// ---------------------------------------------------------------------------
typedef __bf16 bf16_t;
typedef bf16_t v16bf __attribute__((ext_vector_type(16)));
typedef float  v8f   __attribute__((ext_vector_type(8)));
typedef unsigned short u16;
typedef u16 v8u16  __attribute__((ext_vector_type(8)));
typedef u16 v16u16 __attribute__((ext_vector_type(16)));

#define DEVFN static __device__ __forceinline__

// Hardware f32 -> bf16 conversion (fptrunc; RNE on gfx1250)
DEVFN u16 bfbits(float f) {
  return __builtin_bit_cast(u16, (bf16_t)f);
}
// Pack two f32 -> one dword of 2 bf16 (fusable to v_cvt_pk_bf16_f32)
DEVFN unsigned pack2bf(float x, float y) {
  return ((unsigned)bfbits(y) << 16) | (unsigned)bfbits(x);
}

DEVFN float fast_rcp(float x) { return __builtin_amdgcn_rcpf(x); }

// Branch-free fast gelu (tanh approx; tanh via one v_exp_f32 + v_rcp_f32)
DEVFN float gelu_tanh(float x) {
  float x3 = x * x * x;
  float t  = 0.7978845608028654f * (x + 0.044715f * x3);
  float e  = __expf(2.0f * t);
  float th = 1.0f - 2.0f * fast_rcp(e + 1.0f);
  return 0.5f * x * (1.0f + th);
}

DEVFN float sigmoidf(float x) { return fast_rcp(1.0f + __expf(-x)); }

// Assemble a v16bf fragment from two contiguous 16-byte LDS runs.
DEVFN v16bf frag_cat(const bf16_t* lo, const bf16_t* hi) {
  v8u16 a = *(const v8u16*)lo;
  v8u16 b = *(const v8u16*)hi;
  v16u16 c = __builtin_shufflevector(a, b, 0, 1, 2, 3, 4, 5, 6, 7,
                                     8, 9, 10, 11, 12, 13, 14, 15);
  return __builtin_bit_cast(v16bf, c);
}

// ---------------------------------------------------------------------------
// Problem dimensions (fixed by the reference)
// ---------------------------------------------------------------------------
constexpr int Bsz = 16;
constexpr int Nsq = 2048;
constexpr int Hin = 256;
constexpr int Hd  = 256;
constexpr int Pd  = 256;
constexpr int Ld  = 4;
constexpr int Mrows = Bsz * Nsq;            // 32768
constexpr size_t SLOT = (size_t)Mrows * Hd; // floats per activation buffer

// GEMM tiling: 128x64 block tile, 8 waves, each wave 32x32 (2x2 WMMA tiles)
constexpr int BM = 128, BN = 64, BK = 32;
constexpr int LDK = BK + 8;                 // 40 bf16 = 80 B row stride (16B aligned)

// ---------------------------------------------------------------------------
// Generic GEMM: out = [ACCUM? out + : ] scale * (A @ W [+ bias])
// A: M x K (f32 row-major), W: K x Nc (f32 row-major), out: M x Nc
// ---------------------------------------------------------------------------
template<bool BIAS, bool ACCUM>
__global__ void __launch_bounds__(256)
gemm_wmma(const float* __restrict__ A, const float* __restrict__ W,
          const float* __restrict__ bias, float* __restrict__ out,
          float scale, int M, int K, int Nc) {
  __shared__ bf16_t As[BM][LDK];   // row-major, K contiguous
  __shared__ bf16_t Wt[BN][LDK];   // TRANSPOSED: per output col, K contiguous

  const int tid   = threadIdx.x;
  const int lane  = tid & 31;
  const int wave  = tid >> 5;
  const int waveM = wave & 3;   // 0..3
  const int waveN = wave >> 2;  // 0..1
  const int rowBase = blockIdx.x * BM;
  const int colBase = blockIdx.y * BN;

  v8f acc[2][2] = {};

  for (int k0 = 0; k0 < K; k0 += BK) {
    if (k0 + BK < K) {  // prefetch next A k-tile (global_prefetch_b8)
      int r = tid >> 5, c = (tid & 31);
      __builtin_prefetch(&A[(size_t)(rowBase + r * 4) * K + (k0 + BK + c)], 0, 1);
    }
    // ---- stage A tile (BM x BK): float2 loads, packed bf16x2 LDS stores ----
    #pragma unroll
    for (int i = 0; i < (BM * BK) / (256 * 2); ++i) {    // 8 iters
      int e = tid + i * 256;      // pair index
      int r = e >> 4;             // 16 pairs per row
      int c = (e & 15) * 2;
      float2 f = *(const float2*)&A[(size_t)(rowBase + r) * K + (k0 + c)];
      *(unsigned*)&As[r][c] = pack2bf(f.x, f.y);
    }
    // ---- stage W tile (BK x BN) transposed: Wt[col][k], packed pairs ----
    #pragma unroll
    for (int i = 0; i < (BK * BN) / (256 * 2); ++i) {    // 4 iters
      int e = tid + i * 256;
      int c = e & 63;             // col within tile
      int r = (e >> 6) * 2;       // even k within tile
      float w0 = W[(size_t)(k0 + r) * Nc + (colBase + c)];
      float w1 = W[(size_t)(k0 + r + 1) * Nc + (colBase + c)];
      *(unsigned*)&Wt[c][r] = pack2bf(w0, w1);
    }
    __syncthreads();

    // ---- A fragments: lane<16 -> K{0..7,16..23}; lane>=16 -> K{8..15,24..31}
    const int kbA = (lane >> 4) * 8;
    v16bf afrag[2];
    #pragma unroll
    for (int mi = 0; mi < 2; ++mi) {
      int r = waveM * 32 + mi * 16 + (lane & 15);
      afrag[mi] = frag_cat(&As[r][kbA], &As[r][kbA + 16]);
    }
    // ---- B fragments: lane<16 -> K{0..7,16..23}; lane>=16 -> K{8..15,24..31}
    const int kbB = (lane >> 4) * 8;
    v16bf wfrag[2];
    #pragma unroll
    for (int ni = 0; ni < 2; ++ni) {
      int c = waveN * 32 + ni * 16 + (lane & 15);
      wfrag[ni] = frag_cat(&Wt[c][kbB], &Wt[c][kbB + 16]);
    }
    #pragma unroll
    for (int mi = 0; mi < 2; ++mi)
      #pragma unroll
      for (int ni = 0; ni < 2; ++ni)
        acc[mi][ni] = __builtin_amdgcn_wmma_f32_16x16x32_bf16(
            false, afrag[mi], false, wfrag[ni], (short)0, acc[mi][ni],
            false, false);
    __syncthreads();
  }

  // ---- epilogue: C/D layout = VGPR r -> M = (lane>=16? 8:0)+r, N = lane%16
  #pragma unroll
  for (int ni = 0; ni < 2; ++ni) {
    int col = colBase + waveN * 32 + ni * 16 + (lane & 15);
    float bv = BIAS ? bias[col] : 0.0f;
    #pragma unroll
    for (int mi = 0; mi < 2; ++mi) {
      int rbase = rowBase + waveM * 32 + mi * 16 + (lane >> 4) * 8;
      #pragma unroll
      for (int r = 0; r < 8; ++r) {
        float v = (acc[mi][ni][r] + bv) * scale;
        size_t idx = (size_t)(rbase + r) * Nc + col;
        if (ACCUM) out[idx] += v; else out[idx] = v;
      }
    }
  }
}

// ---------------------------------------------------------------------------
// Fused GLU: g = gelu(Y); U += (g@W1 + b1) * sigmoid(g@W2 + b2)
// ---------------------------------------------------------------------------
__global__ void __launch_bounds__(256)
glu_wmma(const float* __restrict__ Y,
         const float* __restrict__ W1, const float* __restrict__ b1,
         const float* __restrict__ W2, const float* __restrict__ b2,
         float* __restrict__ U, int M, int K, int Nc) {
  __shared__ bf16_t As[BM][LDK];
  __shared__ bf16_t W1t[BN][LDK];
  __shared__ bf16_t W2t[BN][LDK];

  const int tid   = threadIdx.x;
  const int lane  = tid & 31;
  const int wave  = tid >> 5;
  const int waveM = wave & 3;
  const int waveN = wave >> 2;
  const int rowBase = blockIdx.x * BM;
  const int colBase = blockIdx.y * BN;

  v8f acc1[2][2] = {};
  v8f acc2[2][2] = {};

  for (int k0 = 0; k0 < K; k0 += BK) {
    #pragma unroll
    for (int i = 0; i < (BM * BK) / (256 * 2); ++i) {
      int e = tid + i * 256;
      int r = e >> 4;
      int c = (e & 15) * 2;
      float2 f = *(const float2*)&Y[(size_t)(rowBase + r) * K + (k0 + c)];
      *(unsigned*)&As[r][c] = pack2bf(gelu_tanh(f.x), gelu_tanh(f.y));
    }
    #pragma unroll
    for (int i = 0; i < (BK * BN) / (256 * 2); ++i) {
      int e = tid + i * 256;
      int c = e & 63;
      int r = (e >> 6) * 2;
      size_t w0i = (size_t)(k0 + r) * Nc + (colBase + c);
      size_t w1i = (size_t)(k0 + r + 1) * Nc + (colBase + c);
      *(unsigned*)&W1t[c][r] = pack2bf(W1[w0i], W1[w1i]);
      *(unsigned*)&W2t[c][r] = pack2bf(W2[w0i], W2[w1i]);
    }
    __syncthreads();

    const int kbA = (lane >> 4) * 8;
    v16bf afrag[2];
    #pragma unroll
    for (int mi = 0; mi < 2; ++mi) {
      int r = waveM * 32 + mi * 16 + (lane & 15);
      afrag[mi] = frag_cat(&As[r][kbA], &As[r][kbA + 16]);
    }
    const int kbB = (lane >> 4) * 8;
    #pragma unroll
    for (int ni = 0; ni < 2; ++ni) {
      int c = waveN * 32 + ni * 16 + (lane & 15);
      v16bf w1f = frag_cat(&W1t[c][kbB], &W1t[c][kbB + 16]);
      v16bf w2f = frag_cat(&W2t[c][kbB], &W2t[c][kbB + 16]);
      #pragma unroll
      for (int mi = 0; mi < 2; ++mi) {
        acc1[mi][ni] = __builtin_amdgcn_wmma_f32_16x16x32_bf16(
            false, afrag[mi], false, w1f, (short)0, acc1[mi][ni], false, false);
        acc2[mi][ni] = __builtin_amdgcn_wmma_f32_16x16x32_bf16(
            false, afrag[mi], false, w2f, (short)0, acc2[mi][ni], false, false);
      }
    }
    __syncthreads();
  }

  #pragma unroll
  for (int ni = 0; ni < 2; ++ni) {
    int col = colBase + waveN * 32 + ni * 16 + (lane & 15);
    float b1v = b1[col];
    float b2v = b2[col];
    #pragma unroll
    for (int mi = 0; mi < 2; ++mi) {
      int rbase = rowBase + waveM * 32 + mi * 16 + (lane >> 4) * 8;
      #pragma unroll
      for (int r = 0; r < 8; ++r) {
        float t1 = acc1[mi][ni][r] + b1v;
        float t2 = acc2[mi][ni][r] + b2v;
        size_t idx = (size_t)(rbase + r) * Nc + col;
        U[idx] += t1 * sigmoidf(t2);
      }
    }
  }
}

// ---------------------------------------------------------------------------
// LayerNorm over last dim (H == blockDim.x == 256), one row per block
// ---------------------------------------------------------------------------
__global__ void __launch_bounds__(256)
layernorm_k(const float* __restrict__ in, const float* __restrict__ sc,
            const float* __restrict__ bi, float* __restrict__ outv, int H) {
  __shared__ float s1[256];
  __shared__ float s2[256];
  const int t = threadIdx.x;
  const size_t row = blockIdx.x;
  float x = in[row * H + t];
  s1[t] = x;
  s2[t] = x * x;
  __syncthreads();
  #pragma unroll
  for (int off = 128; off > 0; off >>= 1) {
    if (t < off) { s1[t] += s1[t + off]; s2[t] += s2[t + off]; }
    __syncthreads();
  }
  float mean = s1[0] * (1.0f / 256.0f);
  float var  = s2[0] * (1.0f / 256.0f) - mean * mean;
  float inv  = rsqrtf(var + 1e-5f);
  outv[row * H + t] = (x - mean) * inv * sc[t] + bi[t];
}

// ---------------------------------------------------------------------------
// Complex diagonal SSM scan along N. One thread per (b,p) channel.
// ---------------------------------------------------------------------------
__global__ void __launch_bounds__(256)
scan_k(const float* __restrict__ dt,
       const float* __restrict__ lam_re_in, const float* __restrict__ lam_im_in,
       const float* __restrict__ log_delta,
       const float* __restrict__ init_re, const float* __restrict__ init_im,
       const float* __restrict__ bu_re, const float* __restrict__ bu_im,
       float* __restrict__ xr_re, float* __restrict__ xr_im,
       float* __restrict__ xl_re, float* __restrict__ xl_im,
       int N, int P) {
  const int b = blockIdx.x;
  const int p = threadIdx.x;
  const float lre = -__expf(lam_re_in[p]);
  const float lim = lam_im_in[p];
  const float eld = __expf(log_delta[p]);
  float xre = init_re[p];
  float xim = init_im[p];
  const size_t base = (size_t)b * N * P + p;
  for (int n = 0; n < N; ++n) {
    float d = dt[(size_t)b * N + n] * eld;
    float mag = __expf(lre * d);
    float s, c;
    __sincosf(lim * d, &s, &c);
    float are = mag * c, aim = mag * s;
    float lr = are * xre - aim * xim;
    float li = are * xim + aim * xre;
    size_t idx = base + (size_t)n * P;
    xl_re[idx] = lr;
    xl_im[idx] = li;
    xre = lr + bu_re[idx];
    xim = li + bu_im[idx];
    xr_re[idx] = xre;
    xr_im[idx] = xim;
  }
}

// ---------------------------------------------------------------------------
// out[b,0,:] = 0 ; out[b,n,:] = left_u[b,n,:] for n >= 1
// ---------------------------------------------------------------------------
__global__ void __launch_bounds__(256)
shift_out_k(const float* __restrict__ left_u, float* __restrict__ out,
            int N, int H) {
  size_t idx = (size_t)blockIdx.x * blockDim.x + threadIdx.x;
  int n = (int)((idx / H) % N);
  out[idx] = (n == 0) ? 0.0f : left_u[idx];
}

// ---------------------------------------------------------------------------
// Launch
// ---------------------------------------------------------------------------
extern "C" void kernel_launch(void* const* d_in, const int* in_sizes, int n_in,
                              void* d_out, int out_size, void* d_ws, size_t ws_size,
                              hipStream_t stream) {
  (void)in_sizes; (void)n_in; (void)out_size; (void)ws_size;

  const float* x_payload = (const float*)d_in[0];   // (B,N,Hin)
  const float* dt        = (const float*)d_in[1];   // (B,N)
  const float* embed_W   = (const float*)d_in[2];   // (Hin,H)
  const float* embed_b   = (const float*)d_in[3];   // (H)
  const float* ln_scale  = (const float*)d_in[4];   // (L,H)
  const float* ln_bias   = (const float*)d_in[5];   // (L,H)
  const float* Lambda_re = (const float*)d_in[6];   // (L,P)
  const float* Lambda_im = (const float*)d_in[7];   // (L,P)
  const float* log_delta = (const float*)d_in[8];   // (L,P)
  const float* B_re      = (const float*)d_in[9];   // (L,H,P)
  const float* B_im      = (const float*)d_in[10];  // (L,H,P)
  const float* C_re      = (const float*)d_in[11];  // (L,P,H)
  const float* C_im      = (const float*)d_in[12];  // (L,P,H)
  const float* W1        = (const float*)d_in[13];  // (L,H,H)
  const float* b1        = (const float*)d_in[14];  // (L,H)
  const float* W2        = (const float*)d_in[15];  // (L,H,H)
  const float* b2        = (const float*)d_in[16];  // (L,H)
  const float* init_re   = (const float*)d_in[17];  // (L,P)
  const float* init_im   = (const float*)d_in[18];  // (L,P)
  float* out = (float*)d_out;

  float* ws = (float*)d_ws;
  float* right_u = ws + 0 * SLOT;
  float* left_u  = ws + 1 * SLOT;
  float* v_r     = ws + 2 * SLOT;
  float* v_l     = ws + 3 * SLOT;
  float* bu_re   = ws + 4 * SLOT;
  float* bu_im   = ws + 5 * SLOT;
  float* xr_re   = ws + 6 * SLOT;
  float* xr_im   = ws + 7 * SLOT;
  float* xl_re   = ws + 8 * SLOT;
  float* xl_im   = ws + 9 * SLOT;
  float* y_r     = ws + 10 * SLOT;
  float* y_l     = ws + 11 * SLOT;

  const dim3 blk(256);
  const dim3 gemm_grid(Mrows / BM, Hd / BN);   // 256 x 4

  // --- embedding: alpha = x @ embed_W + embed_b -> both streams ---
  gemm_wmma<true, false><<<gemm_grid, blk, 0, stream>>>(
      x_payload, embed_W, embed_b, right_u, 1.0f, Mrows, Hin, Hd);
  hipMemcpyAsync(left_u, right_u, SLOT * sizeof(float),
                 hipMemcpyDeviceToDevice, stream);

  for (int l = 0; l < Ld; ++l) {
    const float* lns = ln_scale + (size_t)l * Hd;
    const float* lnb = ln_bias  + (size_t)l * Hd;
    const float* Bre = B_re + (size_t)l * Hd * Pd;
    const float* Bim = B_im + (size_t)l * Hd * Pd;
    const float* Cre = C_re + (size_t)l * Pd * Hd;
    const float* Cim = C_im + (size_t)l * Pd * Hd;
    const float* W1l = W1 + (size_t)l * Hd * Hd;
    const float* W2l = W2 + (size_t)l * Hd * Hd;
    const float* b1l = b1 + (size_t)l * Hd;
    const float* b2l = b2 + (size_t)l * Hd;

    // layernorm both streams
    layernorm_k<<<Mrows, blk, 0, stream>>>(right_u, lns, lnb, v_r, Hd);
    layernorm_k<<<Mrows, blk, 0, stream>>>(left_u,  lns, lnb, v_l, Hd);

    // Bu = v_r @ (B_re + i*B_im)
    gemm_wmma<false, false><<<gemm_grid, blk, 0, stream>>>(
        v_r, Bre, nullptr, bu_re, 1.0f, Mrows, Hd, Pd);
    gemm_wmma<false, false><<<gemm_grid, blk, 0, stream>>>(
        v_r, Bim, nullptr, bu_im, 1.0f, Mrows, Hd, Pd);

    // recurrence along N
    scan_k<<<Bsz, Pd, 0, stream>>>(
        dt, Lambda_re + (size_t)l * Pd, Lambda_im + (size_t)l * Pd,
        log_delta + (size_t)l * Pd, init_re + (size_t)l * Pd,
        init_im + (size_t)l * Pd, bu_re, bu_im,
        xr_re, xr_im, xl_re, xl_im, Nsq, Pd);

    // readout: y = Re(x) @ C_re - Im(x) @ C_im
    gemm_wmma<false, false><<<gemm_grid, blk, 0, stream>>>(
        xr_re, Cre, nullptr, y_r, 1.0f, Mrows, Pd, Hd);
    gemm_wmma<false, true><<<gemm_grid, blk, 0, stream>>>(
        xr_im, Cim, nullptr, y_r, -1.0f, Mrows, Pd, Hd);
    gemm_wmma<false, false><<<gemm_grid, blk, 0, stream>>>(
        xl_re, Cre, nullptr, y_l, 1.0f, Mrows, Pd, Hd);
    gemm_wmma<false, true><<<gemm_grid, blk, 0, stream>>>(
        xl_im, Cim, nullptr, y_l, -1.0f, Mrows, Pd, Hd);

    // residual GLU update on both streams
    glu_wmma<<<gemm_grid, blk, 0, stream>>>(
        y_r, W1l, b1l, W2l, b2l, right_u, Mrows, Hd, Hd);
    glu_wmma<<<gemm_grid, blk, 0, stream>>>(
        y_l, W1l, b1l, W2l, b2l, left_u, Mrows, Hd, Hd);
  }

  // final shifted output
  shift_out_k<<<(unsigned)(SLOT / 256), blk, 0, stream>>>(left_u, out, Nsq, Hd);
}
